// BiLSTMClassifier_15848429322679
// MI455X (gfx1250) — matrix operand order repaired and checked by reading, not compiled, not run on hardware
//
#include <hip/hip_runtime.h>

typedef __attribute__((ext_vector_type(16))) _Float16 v16h;
typedef __attribute__((ext_vector_type(8)))  _Float16 v8h;
typedef __attribute__((ext_vector_type(8)))  float    v8f;

#define B_   64
#define T_   1024
#define D_   256
#define H_   256
#define G4_  1024   /* 4*H */
#define NT_  64     /* G4_/16 N-tiles */

__device__ __forceinline__ float sigmoidf_(float x) {
    return 1.0f / (1.0f + __expf(-x));
}

__device__ __forceinline__ v16h cat16(v8h lo, v8h hi) {
    return __builtin_shufflevector(lo, hi, 0,1,2,3,4,5,6,7,8,9,10,11,12,13,14,15);
}

// ---------------- lengths[b] = count(x[b,t] != 0) ----------------
__global__ void lengths_kernel(const int* __restrict__ x, int* __restrict__ lengths) {
    __shared__ int red[256];
    const int b = blockIdx.x, tid = threadIdx.x;
    int s = 0;
    for (int t = tid; t < T_; t += 256) s += (x[b * T_ + t] != 0) ? 1 : 0;
    red[tid] = s;
    __syncthreads();
    for (int o = 128; o > 0; o >>= 1) {
        if (tid < o) red[tid] += red[tid + o];
        __syncthreads();
    }
    if (tid == 0) lengths[b] = red[0];
}

// ---------------- embedding gather (f32 -> f16 row-major) ----------------
__global__ void embed_kernel(const int* __restrict__ x, const float* __restrict__ E,
                             _Float16* __restrict__ act) {
    const int pos = blockIdx.x;      // b*T + t
    const int d   = threadIdx.x;     // 0..255
    const int tok = x[pos];
    act[(size_t)pos * D_ + d] = (_Float16)E[(size_t)tok * D_ + d];
}

// ---------------- pack W (R x 1024, f32 row-major) into WMMA B-fragment tiles ----
// Tile (kt,nt): 32x16 block. Lane l holds 16 f16: element e <-> K = kt*32 + (e<8 ? (l/16)*8 + e : 16 + (l/16)*8 + e-8),
// N = nt*16 + (l%16). One v16h (32B) per lane, lane-contiguous.
__global__ void pack_kernel(const float* __restrict__ W, _Float16* __restrict__ P) {
    const int kt = blockIdx.x, nt = blockIdx.y, lane = threadIdx.x;
    const int khalf = lane >> 4, lm = lane & 15;
    alignas(32) _Float16 tmp[16];
    for (int e = 0; e < 16; ++e) {
        const int k = kt * 32 + (e < 8 ? khalf * 8 + e : 16 + khalf * 8 + (e - 8));
        const int n = nt * 16 + lm;
        tmp[e] = (_Float16)W[(size_t)k * G4_ + n];
    }
    *(v16h*)(P + ((size_t)kt * NT_ + nt) * 512 + lane * 16) = *(const v16h*)tmp;
}

// ---------------- batched input projection: out = A @ W[ktBegin*32: , :] + bias ----
// A: (M=65536, Kin) f16 row-major.  One 16x16 output tile per wave, K-loop of WMMA.
__global__ void __launch_bounds__(256) gemm_xproj(
        const _Float16* __restrict__ A, int Kin,
        const _Float16* __restrict__ P, int ktBegin, int ktCount,
        const float* __restrict__ bias, float* __restrict__ out) {
    const int wave = threadIdx.x >> 5, lane = threadIdx.x & 31;
    const int hf = lane >> 4, lm = lane & 15;
    const int nt = blockIdx.x;
    const int mt = blockIdx.y * 8 + wave;
    const int row = mt * 16 + lm;

    v8f acc = {};
    for (int kt = 0; kt < ktCount; ++kt) {
        const _Float16* ap = A + (size_t)row * Kin + kt * 32 + hf * 8;
        v8h alo = *(const v8h*)ap;
        v8h ahi = *(const v8h*)(ap + 16);
        v16h a = cat16(alo, ahi);
        v16h bf = *(const v16h*)(P + ((size_t)(ktBegin + kt) * NT_ + nt) * 512 + lane * 16);
        acc = __builtin_amdgcn_wmma_f32_16x16x32_f16(false, a, false, bf,
                                                     (short)0, acc, false, false);
    }
    const int n = nt * 16 + lm;
    const float bv = bias[n];
#pragma unroll
    for (int v = 0; v < 8; ++v) {
        const int m = mt * 16 + hf * 8 + v;
        out[(size_t)m * G4_ + n] = acc[v] + bv;
    }
}

// ---------------- recurrent scan: one workgroup per direction --------------------
// h (64x256 f16) lives in LDS; c lives in VGPRs (one v8f per job, persistent over T).
// 64 jobs = 4 M-tiles x 16 gate-groups; wave w owns jobs 2w, 2w+1. Each job computes
// all 4 gate tiles (f,i,g,o) of its 16x16 (batch,hidden) patch -> A-frag reuse x4 and
// fully in-register LSTM cell update.
__global__ void __launch_bounds__(1024) lstm_scan(
        const _Float16* __restrict__ Pf, const _Float16* __restrict__ Pb,
        const float* __restrict__ Xf, const float* __restrict__ Xb,
        const int* __restrict__ lengths,
        _Float16* __restrict__ outs,        // (B,T,2H) concat, or nullptr
        float* __restrict__ hcat) {         // (B,2H) final states
    const int dir = blockIdx.x;
    const _Float16* P = dir ? Pb : Pf;
    const float* Xp = dir ? Xb : Xf;
    const int dbase = dir ? H_ : 0;

    __shared__ _Float16 hS[B_ * H_];
    __shared__ int lenS[B_];

    const int tid = threadIdx.x;
    for (int i = tid; i < B_ * H_; i += 1024) hS[i] = (_Float16)0.0f;
    if (tid < B_) lenS[tid] = lengths[tid];
    __syncthreads();

    const int wave = tid >> 5, lane = tid & 31;
    const int hf = lane >> 4, lm = lane & 15;
    const v8f vzero = {};
    v8f cst[2];
    cst[0] = vzero;
    cst[1] = vzero;

    for (int t = 0; t < T_; ++t) {
        const int te = dir ? (T_ - 1 - t) : t;
        v8f acc[2][4];

        // ---- compute z_h = h_{t-1} @ Wh for both jobs (reads hS) ----
#pragma unroll
        for (int jj = 0; jj < 2; ++jj) {
            const int job = (wave << 1) | jj;
            const int mt = job & 3, grp = job >> 2;
            acc[jj][0] = vzero; acc[jj][1] = vzero;
            acc[jj][2] = vzero; acc[jj][3] = vzero;
            const int row = mt * 16 + lm;
            for (int kt = 0; kt < 8; ++kt) {
                const _Float16* ap = &hS[row * H_ + kt * 32 + hf * 8];
                v8h alo = *(const v8h*)ap;
                v8h ahi = *(const v8h*)(ap + 16);
                v16h a = cat16(alo, ahi);
#pragma unroll
                for (int gi = 0; gi < 4; ++gi) {
                    const int ntile = gi * 16 + grp;
                    v16h bf = *(const v16h*)(P + ((size_t)kt * NT_ + ntile) * 512 + lane * 16);
                    acc[jj][gi] = __builtin_amdgcn_wmma_f32_16x16x32_f16(
                        false, a, false, bf, (short)0, acc[jj][gi], false, false);
                }
            }
        }
        __syncthreads();   // all reads of h_{t-1} complete

        // ---- gates + cell update; write h_t back into hS ----
#pragma unroll
        for (int jj = 0; jj < 2; ++jj) {
            const int job = (wave << 1) | jj;
            const int mt = job & 3, grp = job >> 2;
            const int n = grp * 16 + lm;
#pragma unroll
            for (int v = 0; v < 8; ++v) {
                const int b = mt * 16 + hf * 8 + v;
                const size_t xb = ((size_t)b * T_ + te) * G4_ + n;
                const float zf = acc[jj][0][v] + Xp[xb + 0 * H_];
                const float zi = acc[jj][1][v] + Xp[xb + 1 * H_];
                const float zg = acc[jj][2][v] + Xp[xb + 2 * H_];
                const float zo = acc[jj][3][v] + Xp[xb + 3 * H_];
                const float fg = sigmoidf_(zf), ig = sigmoidf_(zi);
                const float gg = tanhf(zg),     og = sigmoidf_(zo);
                const float cold = cst[jj][v];
                const float cnew = fg * cold + ig * gg;
                const float hnew = og * tanhf(cnew);
                const float hold = (float)hS[b * H_ + n];
                const bool  msk  = te < lenS[b];
                const float hv   = msk ? hnew : hold;
                cst[jj][v] = msk ? cnew : cold;
                hS[b * H_ + n] = (_Float16)hv;
                if (outs) outs[((size_t)b * T_ + te) * (2 * H_) + dbase + n] = (_Float16)hv;
            }
        }
        __syncthreads();   // h_t visible to all waves
    }

    // final hidden state -> hcat
#pragma unroll
    for (int jj = 0; jj < 2; ++jj) {
        const int job = (wave << 1) | jj;
        const int mt = job & 3, grp = job >> 2;
        const int n = grp * 16 + lm;
#pragma unroll
        for (int v = 0; v < 8; ++v) {
            const int b = mt * 16 + hf * 8 + v;
            hcat[b * (2 * H_) + dbase + n] = (float)hS[b * H_ + n];
        }
    }
}

// ---------------- classifier: out = hcat @ Wfc + bfc (64x512x2) -----------------
__global__ void fc_kernel(const float* __restrict__ hcat, const float* __restrict__ Wfc,
                          const float* __restrict__ bfc, float* __restrict__ out) {
    const int id = threadIdx.x;
    if (id >= B_ * 2) return;
    const int b = id >> 1, c = id & 1;
    float s = bfc[c];
    for (int k = 0; k < 2 * H_; ++k) s += hcat[b * (2 * H_) + k] * Wfc[k * 2 + c];
    out[b * 2 + c] = s;
}

extern "C" void kernel_launch(void* const* d_in, const int* in_sizes, int n_in,
                              void* d_out, int out_size, void* d_ws, size_t ws_size,
                              hipStream_t stream) {
    const int*   x   = (const int*)  d_in[0];
    const float* E   = (const float*)d_in[1];
    const float* Wf0 = (const float*)d_in[2];
    const float* bf0 = (const float*)d_in[3];
    const float* Wb0 = (const float*)d_in[4];
    const float* bb0 = (const float*)d_in[5];
    const float* Wf1 = (const float*)d_in[6];
    const float* bf1 = (const float*)d_in[7];
    const float* Wb1 = (const float*)d_in[8];
    const float* bb1 = (const float*)d_in[9];
    const float* Wfc = (const float*)d_in[10];
    const float* bfc = (const float*)d_in[11];
    float* out = (float*)d_out;

    char* ws = (char*)d_ws;
    size_t off = 0;
    auto take = [&](size_t bytes) -> char* {
        char* p = ws + off;
        off += (bytes + 255) & ~(size_t)255;
        return p;
    };
    int*      lengths = (int*)      take(B_ * 4);
    float*    hcat    = (float*)    take((size_t)B_ * 2 * H_ * 4);
    _Float16* act0    = (_Float16*) take((size_t)B_ * T_ * D_ * 2);
    _Float16* actcat  = (_Float16*) take((size_t)B_ * T_ * 2 * H_ * 2);
    float*    Xf      = (float*)    take((size_t)B_ * T_ * G4_ * 4);
    float*    Xb      = (float*)    take((size_t)B_ * T_ * G4_ * 4);
    _Float16* pWf0    = (_Float16*) take((size_t)512 * G4_ * 2);
    _Float16* pWb0    = (_Float16*) take((size_t)512 * G4_ * 2);
    _Float16* pWf1    = (_Float16*) take((size_t)768 * G4_ * 2);
    _Float16* pWb1    = (_Float16*) take((size_t)768 * G4_ * 2);

    lengths_kernel<<<B_, 256, 0, stream>>>(x, lengths);
    embed_kernel<<<B_ * T_, D_, 0, stream>>>(x, E, act0);

    pack_kernel<<<dim3(16, 64), 32, 0, stream>>>(Wf0, pWf0);
    pack_kernel<<<dim3(16, 64), 32, 0, stream>>>(Wb0, pWb0);
    pack_kernel<<<dim3(24, 64), 32, 0, stream>>>(Wf1, pWf1);
    pack_kernel<<<dim3(24, 64), 32, 0, stream>>>(Wb1, pWb1);

    // ---- layer 0: Xproj = act0 @ Wx + b (Wx = rows 256.. of W); then biLSTM scan
    gemm_xproj<<<dim3(64, 512), 256, 0, stream>>>(act0, D_, pWf0, 8, 8,  bf0, Xf);
    gemm_xproj<<<dim3(64, 512), 256, 0, stream>>>(act0, D_, pWb0, 8, 8,  bb0, Xb);
    lstm_scan<<<2, 1024, 0, stream>>>(pWf0, pWb0, Xf, Xb, lengths, actcat, hcat);

    // ---- layer 1: input = concat(fo,bo); outputs not needed, only final states
    gemm_xproj<<<dim3(64, 512), 256, 0, stream>>>(actcat, 2 * H_, pWf1, 8, 16, bf1, Xf);
    gemm_xproj<<<dim3(64, 512), 256, 0, stream>>>(actcat, 2 * H_, pWb1, 8, 16, bb1, Xb);
    lstm_scan<<<2, 1024, 0, stream>>>(pWf1, pWb1, Xf, Xb, lengths,
                                      (_Float16*)nullptr, hcat);

    fc_kernel<<<1, 128, 0, stream>>>(hcat, Wfc, bfc, out);

    (void)in_sizes; (void)n_in; (void)out_size; (void)ws_size;
}
